// LinearEmbedder_75479755260191
// MI455X (gfx1250) — matrix operand order repaired and checked by reading, not compiled, not run on hardware
//
#include <hip/hip_runtime.h>
#include <hip/hip_bf16.h>
#include <math.h>

typedef __attribute__((ext_vector_type(16))) _Float16 v16h;
typedef __attribute__((ext_vector_type(8)))  _Float16 v8h;
typedef __attribute__((ext_vector_type(8)))  float    v8f;
typedef __attribute__((ext_vector_type(4)))  float    v4f;

#define ENC_DIM 253
#define KPAD    256
#define NOUT    256
#define ROWS    64          // rows per block
#define LDS_STRIDE 264      // 256 + 8-half pad (breaks 512B bank stride)
#define STG_STRIDE 68       // 64 + 4-float pad (bank-spread for staging)

__constant__ int c_tables[4][10] = {
  {0, 4, 1, 5, 3, 6, 7, 8, 9, 10},
  {0, 11, 1, 2, 12, 3, 13, 14, 15, 16},
  {0, 17, 1, 18, 3, 4, 19, 20, 21, 22},
  {0, 23, 1, 2, 5, 24, 25, 26, 27, 28}
};
__constant__ int c_offsets[29] = {
  0, 4, 14, 16, 28, 42, 50, 59, 69, 83, 95, 101, 109, 114, 123,
  135, 147, 152, 163, 171, 177, 184, 198, 202, 213, 221, 229, 237, 245
};

// ---- prologue: W (f32, 256x253) -> f16 padded to K=256 in workspace ----
__global__ __launch_bounds__(256) void w_to_f16(const float* __restrict__ W,
                                                _Float16* __restrict__ Wh) {
  const int o = blockIdx.x;          // 0..255 output channel
  const int k = threadIdx.x;         // 0..255 padded K
  float v = (k < ENC_DIM) ? W[o * ENC_DIM + k] : 0.0f;
  Wh[o * KPAD + k] = (_Float16)v;
}

union AFrag { v16h v; unsigned int u[8]; };
union BFrag { v16h v; v8h h[2]; };

// ---- fused: encode-to-LDS + f16 WMMA GEMM + exact GELU + coalesced NT store ----
__global__ __launch_bounds__(256) void fused_encode_gemm_gelu(
    const int* __restrict__ panose, const _Float16* __restrict__ Wh,
    const float* __restrict__ bias, float* __restrict__ out) {
  __shared__ _Float16 encS[ROWS * LDS_STRIDE];
  __shared__ __attribute__((aligned(16))) float stg[8][16 * STG_STRIDE]; // per-wave slice

  const int tid = threadIdx.x;
  const long rowBase = (long)blockIdx.x * ROWS;

  // 1) zero-fill LDS encoding tile with 16B stores
  {
    v8h zero = {};
    v8h* encV = (v8h*)encS;
    const int nvec = (ROWS * LDS_STRIDE) / 8;   // 2112
    for (int idx = tid; idx < nvec; idx += 256) encV[idx] = zero;
  }
  __syncthreads();

  // 2) sparse one-hot encode: thread t scatters <=10 ones into its row
  if (tid < ROWS) {
    const int* p = panose + (rowBase + tid) * 10;
    const int p0 = __builtin_nontemporal_load(p);
    const int sel = (p0 == 3) ? 1 : ((p0 == 4) ? 2 : ((p0 == 5) ? 3 : 0));
    _Float16* er = encS + tid * LDS_STRIDE;
#pragma unroll
    for (int j = 0; j < 10; ++j) {
      const int v = __builtin_nontemporal_load(p + j);
      const int slot = c_tables[sel][j];
      if (v >= 2) er[c_offsets[slot] + v - 2] = (_Float16)1.0f;
    }
  }
  __syncthreads();

  // 3) WMMA: wave w -> M-tile (w&3), N-half (w>>2); 8 N-tiles each
  const int wave = tid >> 5;
  const int lane = tid & 31;
  const int mt = wave & 3;
  const int nh = wave >> 2;
  const int lg = lane >> 4;   // lane group (0: lanes 0-15, 1: lanes 16-31)
  const int lm = lane & 15;

  v8f acc[8];
#pragma unroll
  for (int nt = 0; nt < 8; ++nt) {
    const float bv = bias[nh * 128 + nt * 16 + lm];  // C layout: N = lane%16
#pragma unroll
    for (int r = 0; r < 8; ++r) acc[nt][r] = bv;     // bias folded into C init
  }

  const _Float16* abase = encS + (mt * 16 + lm) * LDS_STRIDE;  // A: M = lane%16
#pragma unroll
  for (int ks = 0; ks < 8; ++ks) {
    // A fragment per ISA 16-bit A 16x32 layout:
    // VGPR v: K = (v<4?0:16) + 8*lanegrp + 2*(v&3) + half
    AFrag a;
#pragma unroll
    for (int v = 0; v < 8; ++v) {
      const int K = ks * 32 + ((v < 4) ? 0 : 16) + lg * 8 + (v & 3) * 2;
      a.u[v] = *(const unsigned int*)(abase + K);   // merges into 2x ds_load_b128
    }
    // preload all 8 B fragments (one clause; waits drain while WMMAs run)
    const int k0 = ks * 32 + lg * 8;    // B: lanes 0-15 K=0..7,16..23, lanes 16-31 K=8..15,24..31
    BFrag bf[8];
#pragma unroll
    for (int nt = 0; nt < 8; ++nt) {
      const _Float16* wp = Wh + (nh * 128 + nt * 16 + lm) * KPAD + k0;
      bf[nt].h[0] = *(const v8h*)(wp);
      bf[nt].h[1] = *(const v8h*)(wp + 16);
    }
#pragma unroll
    for (int nt = 0; nt < 8; ++nt) {
      acc[nt] = __builtin_amdgcn_wmma_f32_16x16x32_f16(
          /*neg_a=*/false, a.v, /*neg_b=*/false, bf[nt].v,
          /*c_mod=*/(short)0, acc[nt], /*reuse_a=*/false, /*reuse_b=*/false);
    }
  }

  // 4) exact GELU, transpose through wave-private LDS slice, full-line NT stores.
  //    Each half: 4 N-tiles -> 16 rows x 64 cols staged, then 256B-contiguous
  //    float4 stores per half-wave (full 128B cachelines).
  float* mystg = &stg[wave][0];
#pragma unroll
  for (int h = 0; h < 2; ++h) {
#pragma unroll
    for (int t = 0; t < 4; ++t) {
      const int nt = h * 4 + t;
#pragma unroll
      for (int r = 0; r < 8; ++r) {
        const float x = acc[nt][r];
        const float y = 0.5f * x * (1.0f + erff(x * 0.70710678118654752f));
        const int row = lg * 8 + r;          // C layout: M = r + 8*lanegrp
        const int col = t * 16 + lm;         //            N = lane%16
        mystg[row * STG_STRIDE + col] = y;
      }
    }
    // wave-private: in-order LDS ops, no barrier needed
#pragma unroll
    for (int i = 0; i < 8; ++i) {
      const int c = lane + i * 32;           // 256 float4-chunks / 32 lanes
      const int row = c >> 4;                // 0..15
      const int c4 = c & 15;                 // 0..15 (x4 floats)
      const v4f val = *(const v4f*)(mystg + row * STG_STRIDE + c4 * 4);
      const long orow = rowBase + mt * 16 + row;
      float* op = out + orow * NOUT + nh * 128 + h * 64 + c4 * 4;
      __builtin_nontemporal_store(val, (v4f*)op);
    }
  }
}

extern "C" void kernel_launch(void* const* d_in, const int* in_sizes, int n_in,
                              void* d_out, int out_size, void* d_ws, size_t ws_size,
                              hipStream_t stream) {
  (void)n_in; (void)out_size; (void)ws_size;
  const int*   panose = (const int*)d_in[0];   // (B, 10) int32
  const float* W      = (const float*)d_in[1]; // (256, 253) f32
  const float* bias   = (const float*)d_in[2]; // (256,) f32
  float*       out    = (float*)d_out;         // (B, 256) f32
  _Float16*    Wh     = (_Float16*)d_ws;       // 256*256 f16 = 128 KB scratch

  w_to_f16<<<NOUT, 256, 0, stream>>>(W, Wh);

  const int rows = in_sizes[0] / 10;           // B = 262144
  fused_encode_gemm_gelu<<<rows / ROWS, 256, 0, stream>>>(panose, Wh, bias, out);
}
